// SparseMoE_65438121722272
// MI455X (gfx1250) — matrix operand order repaired and checked by reading, not compiled, not run on hardware
//
#include <hip/hip_runtime.h>
#include <hip/hip_bf16.h>

// ---------------------------------------------------------------------------
// Sparse MoE (top-2 of 8 experts), B=4 S=2048 D=1024 H=4096.
// Routed bf16-WMMA implementation:
//   1) zero out + counters
//   2) convert x -> bf16 image
//   3) swizzle W1/W2 -> bf16 WMMA B-fragment layout (1KB per 16x32 fragment)
//   4) router builds per-expert (token, gate) lists
//   5) expert FFN: M=32 token tiles, dual 16x16x32 bf16 WMMA per B-fragment
//      (2x arithmetic intensity vs M=16), split-H, f32 atomic scatter
// ---------------------------------------------------------------------------

#define NTOK 8192      // B*S tokens
#define Dm   1024
#define Em   8
#define Hm   4096
#define HC   512       // H chunk per FFN block
#define MT   32        // token tile (2 WMMA M-tiles sharing each B-fragment)

constexpr int XS = Dm + 8;   // bf16 row stride: 16B-aligned, bank-stride 4 dw
constexpr int HS = HC + 8;
constexpr size_t SMEM_BYTES = (size_t)MT * XS * 2   // XT: gathered x tile
                            + (size_t)MT * HS * 2   // HB: h chunk
                            + (size_t)MT * 4        // gates
                            + (size_t)MT * 4;       // row offsets

typedef __attribute__((ext_vector_type(16))) __bf16 bf16x16;
typedef __attribute__((ext_vector_type(8)))  __bf16 bf16x8;
typedef __attribute__((ext_vector_type(4)))  __bf16 bf16x4;
typedef __attribute__((ext_vector_type(8)))  float  f32x8;

union ABFrag { bf16x16 v; bf16x8 h[2]; };

__device__ __forceinline__ __bf16 to_bf16(float f) {
  union { float f; unsigned u; } in; in.f = f;
  unsigned u = in.u;
  unsigned r = u + 0x7FFFu + ((u >> 16) & 1u);   // round-to-nearest-even
  union { unsigned short s; __bf16 b; } o; o.s = (unsigned short)(r >> 16);
  return o.b;
}

// ---------------------------------------------------------------------------
// Kernel 1: zero output (atomic-accumulated) + per-expert counters
// ---------------------------------------------------------------------------
__global__ __launch_bounds__(256) void zero_kernel(float* __restrict__ out,
                                                   int n4, int* __restrict__ counts) {
  float4 z = {0.f, 0.f, 0.f, 0.f};
  float4* o = (float4*)out;
  for (int i = blockIdx.x * blockDim.x + threadIdx.x; i < n4;
       i += gridDim.x * blockDim.x)
    o[i] = z;
  if (blockIdx.x == 0 && threadIdx.x < Em) counts[threadIdx.x] = 0;
}

// ---------------------------------------------------------------------------
// Kernel 2: x (f32) -> bf16 image, elementwise, float4 in / 8B out
// ---------------------------------------------------------------------------
__global__ __launch_bounds__(256) void convert_x_kernel(const float* __restrict__ x,
                                                        __bf16* __restrict__ xb, int n4) {
  const float4* xi = (const float4*)x;
  bf16x4*       xo = (bf16x4*)xb;
  for (int i = blockIdx.x * blockDim.x + threadIdx.x; i < n4;
       i += gridDim.x * blockDim.x) {
    float4 v = xi[i];
    bf16x4 b = { to_bf16(v.x), to_bf16(v.y), to_bf16(v.z), to_bf16(v.w) };
    xo[i] = b;
  }
}

// ---------------------------------------------------------------------------
// Kernel 3: swizzle W (f32, [K][N] row-major, per expert) into bf16 WMMA
// B-fragment order. Fragment (nt, kb) covers N = nt*16..+15, K = kb*32..+31.
// 1KB layout: elems[half(0..1)][lane(0..31)][8 bf16], where lane -> n = nt*16
// + (lane&15), K = kb*32 + 8*(lane>>4) + half*16 + j.  One wave per fragment.
// ---------------------------------------------------------------------------
__global__ __launch_bounds__(256) void swizzle_w_kernel(
    const float* __restrict__ W, __bf16* __restrict__ WF, int Ktot, int Ntot) {
  const int nt = blockIdx.x;
  const int e  = blockIdx.z;
  const int wv = threadIdx.x >> 5, ln = threadIdx.x & 31;
  const int kb = blockIdx.y * 8 + wv;
  const int nkb = Ktot >> 5;

  const float* We  = W  + (size_t)e * Ktot * Ntot;
  __bf16*      WFe = WF + (size_t)e * Ktot * Ntot;

  const int n  = nt * 16 + (ln & 15);
  const int k0 = kb * 32 + (ln >> 4) * 8;
  const float* sp = We + (size_t)k0 * Ntot + n;       // 16 lanes coalesce on n

  bf16x8 lo, hi;
#pragma unroll
  for (int j = 0; j < 8; ++j) lo[j] = to_bf16(sp[(size_t)j * Ntot]);
#pragma unroll
  for (int j = 0; j < 8; ++j) hi[j] = to_bf16(sp[(size_t)(16 + j) * Ntot]);

  __bf16* dp = WFe + ((size_t)nt * nkb + kb) * 512;   // 512 bf16 = 1KB frag
  *(bf16x8*)(dp + ln * 8)       = lo;                 // coalesced 512B
  *(bf16x8*)(dp + 256 + ln * 8) = hi;                 // coalesced 512B
}

// ---------------------------------------------------------------------------
// Kernel 4: router — one wave32 per token. logits = x·Wg + bg, top-2 softmax,
// atomic append (token, gate) into per-expert compact lists.
// ---------------------------------------------------------------------------
__global__ __launch_bounds__(256) void router_kernel(
    const float* __restrict__ x, const float* __restrict__ Wg,
    const float* __restrict__ bg, int* __restrict__ counts,
    int* __restrict__ tok_ids, float* __restrict__ gate_w) {
  __shared__ float wg[Dm * 9];                 // stride 9 -> bank-conflict free
  const int tid = threadIdx.x;
  for (int i = tid; i < Dm * Em; i += 256) {
    int d = i >> 3, e = i & 7;
    wg[d * 9 + e] = Wg[i];
  }
  __syncthreads();

  const int wv  = tid >> 5;
  const int ln  = tid & 31;
  const int tok = blockIdx.x * 8 + wv;

  float acc[Em];
#pragma unroll
  for (int e = 0; e < Em; ++e) acc[e] = 0.f;

  const float* xr = x + (size_t)tok * Dm;
  for (int d = ln; d < Dm; d += 32) {          // coalesced across lanes
    float xv = xr[d];
#pragma unroll
    for (int e = 0; e < Em; ++e) acc[e] += xv * wg[d * 9 + e];
  }
#pragma unroll
  for (int off = 16; off > 0; off >>= 1)
#pragma unroll
    for (int e = 0; e < Em; ++e) acc[e] += __shfl_xor(acc[e], off);

  if (ln == 0) {
    float lg[Em];
#pragma unroll
    for (int e = 0; e < Em; ++e) lg[e] = acc[e] + bg[e];
    int e1 = 0;
#pragma unroll
    for (int e = 1; e < Em; ++e) if (lg[e] > lg[e1]) e1 = e;
    int e2 = (e1 == 0) ? 1 : 0;
#pragma unroll
    for (int e = 0; e < Em; ++e) if (e != e1 && lg[e] > lg[e2]) e2 = e;
    float g1 = 1.f / (1.f + __expf(lg[e2] - lg[e1]));  // 2-way softmax
    float g2 = 1.f - g1;
    int s1 = atomicAdd(&counts[e1], 1);
    tok_ids[e1 * NTOK + s1] = tok; gate_w[e1 * NTOK + s1] = g1;
    int s2 = atomicAdd(&counts[e2], 1);
    tok_ids[e2 * NTOK + s2] = tok; gate_w[e2 * NTOK + s2] = g2;
  }
}

// ---------------------------------------------------------------------------
// Kernel 5: expert FFN. grid = (token-tile, H-chunk, expert), 256 thr = 8 waves.
// M=32: each B-fragment (1KB, 2x global_load_b128) feeds TWO WMMAs (row tiles
// 0-15 and 16-31) -> 32 bf16-FLOP per L2 byte.
// Phase A: h = relu(x_tile · W1[:, chunk] + b1)   (WMMA bf16, f32 acc)
// Phase B: y += h · W2[chunk, :]                  (WMMA bf16, f32 acc)
// Scatter: out[tok] += gate * (y + b2 once)       (global_atomic_add_f32)
// ---------------------------------------------------------------------------
__global__ __launch_bounds__(256) void moe_ffn_kernel(
    const __bf16* __restrict__ xb,  const __bf16* __restrict__ W1F,
    const float* __restrict__ b1,   const __bf16* __restrict__ W2F,
    const float* __restrict__ b2,   float* __restrict__ out,
    const int* __restrict__ counts, const int* __restrict__ tok_ids,
    const float* __restrict__ gate_w) {
  const int e   = blockIdx.z;
  const int hc  = blockIdx.y;
  const int mt  = blockIdx.x;
  const int cnt = counts[e];
  if (mt * MT >= cnt) return;                  // past this expert's token list

  extern __shared__ char smem[];
  __bf16* XT     = (__bf16*)smem;              // [MT][XS] gathered x tile
  __bf16* HB     = XT + MT * XS;               // [MT][HS] h chunk
  float*  s_gate = (float*)(HB + MT * HS);     // [MT]
  int*    s_row  = (int*)(s_gate + MT);        // [MT] tok * Dm

  const int tid = threadIdx.x;
  if (tid < MT) {
    int idx  = mt * MT + tid;
    bool v   = idx < cnt;
    int tok  = v ? tok_ids[e * NTOK + idx] : 0;
    s_row[tid]  = tok * Dm;
    s_gate[tid] = v ? gate_w[e * NTOK + idx] : 0.f;    // 0 => dead row
  }
  __syncthreads();

  {                                            // gather x rows (bf16) -> LDS
    int row = tid >> 3, c0 = tid & 7;          // 8 threads per row
    const uint4* xr = (const uint4*)(xb + s_row[row]); // 2KB rows, 16B aligned
    uint4* xt = (uint4*)(&XT[row * XS]);
#pragma unroll
    for (int c = c0; c < Dm / 8; c += 8)       // b128 copies, coalesced
      xt[c] = xr[c];
  }
  __syncthreads();

  const int w    = tid >> 5, ln = tid & 31;
  const int m    = ln & 15;                    // A row (tile 0); tile 1 = m+16
  const int kh   = (ln >> 4) * 8;              // A K-half offset (ISA layout)
  const int nlow = ln & 15;                    // C column within tile
  const __bf16* W1Fe = W1F + (size_t)e * Dm * Hm;
  const __bf16* W2Fe = W2F + (size_t)e * Dm * Hm;

  // ---------------- Phase A: x[32,1024] @ W1[1024, HC] -> h[32, HC] ----------
  f32x8 accA[2][4];
#pragma unroll
  for (int s = 0; s < 2; ++s)
#pragma unroll
    for (int t = 0; t < 4; ++t)
#pragma unroll
      for (int r = 0; r < 8; ++r) accA[s][t][r] = 0.f;

  for (int kb = 0; kb < Dm / 32; ++kb) {
    ABFrag a0, a1;
    const __bf16* ap0 = &XT[m * XS + kb * 32 + kh];
    a0.h[0] = *(const bf16x8*)ap0;             // ds_load_b128
    a0.h[1] = *(const bf16x8*)(ap0 + 16);
    const __bf16* ap1 = ap0 + 16 * XS;         // rows 16..31
    a1.h[0] = *(const bf16x8*)ap1;
    a1.h[1] = *(const bf16x8*)(ap1 + 16);
#pragma unroll
    for (int t = 0; t < 4; ++t) {
      int nt = hc * (HC / 16) + w * 4 + t;
      const __bf16* fp = W1Fe + ((size_t)nt * (Dm / 32) + kb) * 512 + ln * 8;
      __builtin_prefetch(fp + 512, 0, 1);      // next k-block fragment
      ABFrag b;
      b.h[0] = *(const bf16x8*)fp;             // coalesced global_load_b128
      b.h[1] = *(const bf16x8*)(fp + 256);
      accA[0][t] = __builtin_amdgcn_wmma_f32_16x16x32_bf16(
          false, a0.v, false, b.v, (short)0, accA[0][t], false, false);
      accA[1][t] = __builtin_amdgcn_wmma_f32_16x16x32_bf16(
          false, a1.v, false, b.v, (short)0, accA[1][t], false, false);
    }
  }

  const int mrow0 = (ln >> 4) * 8;             // C/D: VGPR r -> M = r + mrow0
#pragma unroll
  for (int t = 0; t < 4; ++t) {                // bias + relu -> bf16 h in LDS
    int ncol = w * 64 + t * 16 + nlow;
    float bias = b1[e * Hm + hc * HC + ncol];
#pragma unroll
    for (int s = 0; s < 2; ++s)
#pragma unroll
      for (int r = 0; r < 8; ++r) {
        float hv = accA[s][t][r] + bias;
        HB[(s * 16 + mrow0 + r) * HS + ncol] = to_bf16(hv > 0.f ? hv : 0.f);
      }
  }
  __syncthreads();

  // ---------------- Phase B: h[32, HC] @ W2[HC, 1024] -> y[32, 1024] ---------
  f32x8 accB[2][8];
#pragma unroll
  for (int s = 0; s < 2; ++s)
#pragma unroll
    for (int t = 0; t < 8; ++t)
#pragma unroll
      for (int r = 0; r < 8; ++r) accB[s][t][r] = 0.f;

  const int kb0 = (hc * HC) >> 5;              // chunk base in W2 K-blocks
  for (int kb = 0; kb < HC / 32; ++kb) {
    ABFrag a0, a1;
    const __bf16* ap0 = &HB[m * HS + kb * 32 + kh];
    a0.h[0] = *(const bf16x8*)ap0;
    a0.h[1] = *(const bf16x8*)(ap0 + 16);
    const __bf16* ap1 = ap0 + 16 * HS;
    a1.h[0] = *(const bf16x8*)ap1;
    a1.h[1] = *(const bf16x8*)(ap1 + 16);
#pragma unroll
    for (int t = 0; t < 8; ++t) {
      int nt = w * 8 + t;
      const __bf16* fp = W2Fe + ((size_t)nt * (Hm / 32) + kb0 + kb) * 512 + ln * 8;
      __builtin_prefetch(fp + 512, 0, 1);
      ABFrag b;
      b.h[0] = *(const bf16x8*)fp;
      b.h[1] = *(const bf16x8*)(fp + 256);
      accB[0][t] = __builtin_amdgcn_wmma_f32_16x16x32_bf16(
          false, a0.v, false, b.v, (short)0, accB[0][t], false, false);
      accB[1][t] = __builtin_amdgcn_wmma_f32_16x16x32_bf16(
          false, a1.v, false, b.v, (short)0, accB[1][t], false, false);
    }
  }

  // ---------------- gate-scaled atomic scatter into out ----------------------
#pragma unroll
  for (int t = 0; t < 8; ++t) {
    int ncol = w * 128 + t * 16 + nlow;
    float bias2 = (hc == 0) ? b2[e * Dm + ncol] : 0.f;  // add b2 exactly once
#pragma unroll
    for (int s = 0; s < 2; ++s)
#pragma unroll
      for (int r = 0; r < 8; ++r) {
        int mr = s * 16 + mrow0 + r;
        float g = s_gate[mr];
        if (g != 0.f)
          unsafeAtomicAdd(&out[(size_t)s_row[mr] + ncol],
                          (accB[s][t][r] + bias2) * g);
      }
  }
}

// ---------------------------------------------------------------------------
extern "C" void kernel_launch(void* const* d_in, const int* in_sizes, int n_in,
                              void* d_out, int out_size, void* d_ws, size_t ws_size,
                              hipStream_t stream) {
  (void)in_sizes; (void)n_in; (void)ws_size;
  const float* x  = (const float*)d_in[0];
  const float* Wg = (const float*)d_in[1];
  const float* bg = (const float*)d_in[2];
  const float* W1 = (const float*)d_in[3];
  const float* b1 = (const float*)d_in[4];
  const float* W2 = (const float*)d_in[5];
  const float* b2 = (const float*)d_in[6];
  float* out = (float*)d_out;

  // workspace layout (~145 MB):
  //   [counts 8i][pad][tok_ids E*N i32][gates E*N f32] | xb 16MB | W1F 64MB | W2F 64MB
  constexpr size_t OFF_TOK  = 256;
  constexpr size_t OFF_GATE = OFF_TOK + (size_t)Em * NTOK * sizeof(int);
  constexpr size_t OFF_XB   = 1u << 20;
  constexpr size_t OFF_W1F  = OFF_XB  + (size_t)NTOK * Dm * 2;
  constexpr size_t OFF_W2F  = OFF_W1F + (size_t)Em * Dm * Hm * 2;

  int*    counts  = (int*)d_ws;
  int*    tok_ids = (int*)((char*)d_ws + OFF_TOK);
  float*  gate_w  = (float*)((char*)d_ws + OFF_GATE);
  __bf16* xb      = (__bf16*)((char*)d_ws + OFF_XB);
  __bf16* W1F     = (__bf16*)((char*)d_ws + OFF_W1F);
  __bf16* W2F     = (__bf16*)((char*)d_ws + OFF_W2F);

  zero_kernel<<<2048, 256, 0, stream>>>(out, out_size / 4, counts);
  convert_x_kernel<<<2048, 256, 0, stream>>>(x, xb, NTOK * Dm / 4);
  // W1: K=Dm, N=Hm  -> nt=Hm/16=256, kb=Dm/32=32 (8 waves/block -> y=4)
  swizzle_w_kernel<<<dim3(Hm / 16, (Dm / 32) / 8, Em), 256, 0, stream>>>(W1, W1F, Dm, Hm);
  // W2: K=Hm, N=Dm  -> nt=Dm/16=64, kb=Hm/32=128 (y=16)
  swizzle_w_kernel<<<dim3(Dm / 16, (Hm / 32) / 8, Em), 256, 0, stream>>>(W2, W2F, Hm, Dm);
  router_kernel<<<NTOK / 8, 256, 0, stream>>>(x, Wg, bg, counts, tok_ids, gate_w);
  moe_ffn_kernel<<<dim3(NTOK / MT, Hm / HC, Em), 256, SMEM_BYTES, stream>>>(
      xb, W1F, b1, W2F, b2, out, counts, tok_ids, gate_w);
}
